// DateParser_73358041416073
// MI455X (gfx1250) — matrix operand order repaired and checked by reading, not compiled, not run on hardware
//
#include <hip/hip_runtime.h>

#define B_SZ 1024
#define TXX 512
#define TYY 32
#define NA 64
#define NS 128
#define VIN 64
#define VOUT 32

typedef __bf16 bf16;
typedef bf16  v16bf __attribute__((ext_vector_type(16)));
typedef float v8f   __attribute__((ext_vector_type(8)));

// One lane's worth of a 16x16x32 bf16 WMMA operand: 16 bf16 = 32 bytes.
union __align__(32) Frag { v16bf v; unsigned short u[16]; };

__device__ __forceinline__ unsigned short f2bf(float f) {
    unsigned u = __float_as_uint(f);
    u += 0x7FFFu + ((u >> 16) & 1u);           // round-to-nearest-even
    return (unsigned short)(u >> 16);
}

// fast transcendentals: v_exp_f32 + v_rcp_f32 (no IEEE div fixup chain)
__device__ __forceinline__ float fast_rcp(float x)     { return __builtin_amdgcn_rcpf(x); }
__device__ __forceinline__ float fast_sigmoid(float x) { return fast_rcp(1.0f + __expf(-x)); }
__device__ __forceinline__ float fast_tanh(float x)    { return 2.0f * fast_rcp(1.0f + __expf(-2.0f * x)) - 1.0f; }

// Swizzled A-operand element index for (row m, col k) inside a 16xK tile.
__device__ __forceinline__ int aidx(int m, int k) {
    int kt = k >> 5, kk = k & 31;
    return ((kt * 32) + (m + ((kk >> 4) << 4))) * 16 + (kk & 15);
}

// Fragment-major accumulator index for logical (m, n): ntile*256 + lane*8 + r.
__device__ __forceinline__ int zidx(int m, int n) {
    return ((n >> 4) * 256) + (((n & 15) | ((m >> 3) << 4)) * 8) + (m & 7);
}

__device__ __forceinline__ v16bf loadFrag(const Frag* base, int frag, int lane) {
    return base[frag * 32 + lane].v;
}

__device__ __forceinline__ v8f wmma_bf16(v16bf a, v16bf b, v8f c) {
    return __builtin_amdgcn_wmma_f32_16x16x32_bf16(false, a, false, b, (short)0, c, false, false);
}

// CDNA5 async global->LDS copy (ASYNCcnt-tracked), 16B per lane.
__device__ __forceinline__ void async_b128(unsigned lds_addr, const void* sbase, unsigned voff) {
    asm volatile("global_load_async_to_lds_b128 %0, %1, %2"
                 :: "v"(lds_addr), "v"(voff), "s"(sbase) : "memory");
}
__device__ __forceinline__ void wait_async0() {
    asm volatile("s_wait_asynccnt 0" ::: "memory");
}

// ---------- prep kernels ----------
// W: [N][K] fp32 row-major -> bf16 B-operand fragments, fragment-major.
__global__ void k_convT_sw(const float* __restrict__ W, unsigned short* __restrict__ out,
                           int N, int K) {
    int i = blockIdx.x * blockDim.x + threadIdx.x;
    if (i >= N * K) return;
    int n = i / K, k = i - n * K;
    int kt = k >> 5, kk = k & 31;
    int nt = n >> 4, nn = n & 15;
    int lane = nn + ((kk >> 4) << 4);
    int e = kk & 15;
    int nTiles = N >> 4;
    out[(size_t)((kt * nTiles + nt) * 32 + lane) * 16 + e] = f2bf(W[i]);
}
__global__ void k_addb(const float* __restrict__ a, const float* __restrict__ b,
                       float* __restrict__ o, int n) {
    int i = blockIdx.x * blockDim.x + threadIdx.x;
    if (i < n) o[i] = a[i] + b[i];
}
__global__ void k_zero(float* __restrict__ p, int n) {
    int i = blockIdx.x * blockDim.x + threadIdx.x;
    if (i < n) p[i] = 0.0f;
}

// ---------- encoder: fused scan, z=[X|H]@[Wih;Whh]^T via bf16 WMMA ----------
__global__ __launch_bounds__(256) void k_encoder(
    const float* __restrict__ X,
    const unsigned short* __restrict__ WihT_all,   // [2][2*16 frags][32][16] bf16
    const unsigned short* __restrict__ WhhT_all,   // [2][2*16 frags][32][16] bf16
    const float* __restrict__ bias_all,            // [2][4*NA]
    float* __restrict__ pre_out)                   // [B][TX][NS]
{
    __shared__ Frag sWih[32 * 32];                 // 32 KB (2 ktiles x 16 ntiles)
    __shared__ Frag sWhh[32 * 32];                 // 32 KB
    __shared__ Frag sX[2 * 32];                    // 2 KB  (16 x 64 bf16, swizzled)
    __shared__ Frag sH[2 * 32];                    // 2 KB
    __shared__ float4 sStage[2][16 * VIN / 4];     // 8 KB fp32 double-buffer
    __shared__ float sC[16 * NA];                  // 4 KB
    __shared__ __align__(32) float sZ[16 * 256];   // 16 KB fragment-major accum

    const int tid  = threadIdx.x;
    const int lane = tid & 31;
    const int wave = tid >> 5;
    const int dir  = blockIdx.y;
    const int b0   = blockIdx.x * 16;
    const int G    = 4 * NA;

    // stage weights (already fragment-major in global)
    {
        const uint4* g1 = (const uint4*)(WihT_all + (size_t)dir * VIN * G);
        const uint4* g2 = (const uint4*)(WhhT_all + (size_t)dir * NA * G);
        uint4* d1 = (uint4*)sWih;
        uint4* d2 = (uint4*)sWhh;
        for (int i = tid; i < 2048; i += 256) { d1[i] = g1[i]; d2[i] = g2[i]; }
    }
    unsigned short* sXu = (unsigned short*)sX;
    unsigned short* sHu = (unsigned short*)sH;
    for (int i = tid; i < 16 * NA; i += 256) { sHu[i] = 0; sC[i] = 0.0f; }
    __syncthreads();

    const float* bias = bias_all + dir * G;
    const int m_cp = tid >> 4, q_cp = tid & 15;    // async-copy mapping: 16B per thread

    // prologue: async-stage X tile for t=0
    {
        const int tt0 = dir ? (TXX - 1) : 0;
        unsigned voff = (unsigned)((b0 + m_cp) * TXX + tt0) * 256u + q_cp * 16u;
        async_b128((unsigned)(uintptr_t)&sStage[0][tid], X, voff);
    }

    for (int t = 0; t < TXX; ++t) {
        const int buf = t & 1;
        const int tt  = dir ? (TXX - 1 - t) : t;

        wait_async0();
        __syncthreads();                           // stage[buf] ready for all waves

        // convert fp32 stage -> swizzled bf16 A operand
        const float* stg = (const float*)sStage[buf];
        for (int i = tid; i < 16 * VIN; i += 256) {
            int m = i >> 6, k = i & 63;
            sXu[aidx(m, k)] = f2bf(stg[i]);
        }
        __syncthreads();

        // overlap next tile's copy with WMMA + gate phases
        if (t + 1 < TXX) {
            const int tn = dir ? (TXX - 2 - t) : (t + 1);
            unsigned voff = (unsigned)((b0 + m_cp) * TXX + tn) * 256u + q_cp * 16u;
            async_b128((unsigned)(uintptr_t)&sStage[buf ^ 1][tid], X, voff);
        }

        // z = X@Wih^T + H@Whh^T + bias  (each wave: 2 ntiles x 4 WMMA)
        #pragma unroll
        for (int j = 0; j < 2; ++j) {
            const int nt = wave * 2 + j;
            v8f acc;
            const float bv = bias[nt * 16 + (lane & 15)];
            #pragma unroll
            for (int r = 0; r < 8; ++r) acc[r] = bv;
            acc = wmma_bf16(loadFrag(sX, 0, lane), loadFrag(sWih, 0 * 16 + nt, lane), acc);
            acc = wmma_bf16(loadFrag(sX, 1, lane), loadFrag(sWih, 1 * 16 + nt, lane), acc);
            acc = wmma_bf16(loadFrag(sH, 0, lane), loadFrag(sWhh, 0 * 16 + nt, lane), acc);
            acc = wmma_bf16(loadFrag(sH, 1, lane), loadFrag(sWhh, 1 * 16 + nt, lane), acc);
            *(v8f*)&sZ[nt * 256 + lane * 8] = acc;
        }
        __syncthreads();

        // gates + recurrent state update (fast trans, no IEEE div chain)
        for (int i = tid; i < 16 * NA; i += 256) {
            int m = i >> 6, jj = i & 63;
            float zi = sZ[zidx(m, jj)];
            float zf = sZ[zidx(m, NA + jj)];
            float zg = sZ[zidx(m, 2 * NA + jj)];
            float zo = sZ[zidx(m, 3 * NA + jj)];
            float cc = fast_sigmoid(zf) * sC[i] + fast_sigmoid(zi) * fast_tanh(zg);
            float hh = fast_sigmoid(zo) * fast_tanh(cc);
            sC[i] = cc;
            sHu[aidx(m, jj)] = f2bf(hh);
            pre_out[((size_t)(b0 + m) * TXX + tt) * NS + dir * NA + jj] = hh;
        }
        __syncthreads();
    }
}

// ---------- Ep = pre_out @ W1_pre^T (step-invariant attention term) ----------
__global__ __launch_bounds__(256) void k_ep(
    const float* __restrict__ pre_out, const float* __restrict__ W1, float* __restrict__ Ep)
{
    __shared__ float sW[10 * NS];
    for (int i = threadIdx.x; i < 10 * NS; i += 256) {
        int j = i / NS, k = i - j * NS;
        sW[i] = W1[j * (2 * NS) + NS + k];
    }
    __syncthreads();
    size_t idx = (size_t)blockIdx.x * blockDim.x + threadIdx.x;  // over B*TX
    if (idx >= (size_t)B_SZ * TXX) return;
    const float* po = pre_out + idx * NS;
    float acc[10];
    #pragma unroll
    for (int j = 0; j < 10; ++j) acc[j] = 0.0f;
    for (int k = 0; k < NS; k += 4) {
        float4 v = *(const float4*)(po + k);
        #pragma unroll
        for (int j = 0; j < 10; ++j) {
            const float* w = sW + j * NS + k;
            acc[j] += v.x * w[0] + v.y * w[1] + v.z * w[2] + v.w * w[3];
        }
    }
    float* e = Ep + idx * 10;
    #pragma unroll
    for (int j = 0; j < 10; ++j) e[j] = acc[j];
}

// ---------- per-step attention: scores, softmax over TX, context ----------
__global__ __launch_bounds__(256) void k_attn(
    const float* __restrict__ Ep, const float* __restrict__ pre_out,
    const float* __restrict__ s, const float* __restrict__ W1,
    const float* __restrict__ b1, const float* __restrict__ W2,
    const float* __restrict__ b2, float* __restrict__ ctx)
{
    __shared__ float sU[10];
    __shared__ float sW2[10];
    __shared__ float sScore[TXX];
    __shared__ float sRed[256];
    const int tid = threadIdx.x;
    const int b   = blockIdx.x;

    if (tid < 10) {
        float acc = b1[tid];
        const float* sv = s + (size_t)b * NS;
        const float* w  = W1 + tid * (2 * NS);   // state half: cols [0,NS)
        for (int k = 0; k < NS; ++k) acc += sv[k] * w[k];
        sU[tid]  = acc;
        sW2[tid] = W2[tid];
    }
    __syncthreads();

    const float b2v = b2[0];
    float sc[2];
    float mx = -1e30f;
    #pragma unroll
    for (int q = 0; q < 2; ++q) {
        int t = tid + q * 256;
        const float* ep = Ep + ((size_t)b * TXX + t) * 10;
        float e = b2v;
        #pragma unroll
        for (int j = 0; j < 10; ++j) e += sW2[j] * fast_tanh(ep[j] + sU[j]);
        e = fmaxf(e, 0.0f);
        sc[q] = e;
        mx = fmaxf(mx, e);
    }
    sRed[tid] = mx;
    __syncthreads();
    for (int off = 128; off > 0; off >>= 1) {
        if (tid < off) sRed[tid] = fmaxf(sRed[tid], sRed[tid + off]);
        __syncthreads();
    }
    mx = sRed[0];
    __syncthreads();

    float sum = 0.0f;
    #pragma unroll
    for (int q = 0; q < 2; ++q) {
        float p = __expf(sc[q] - mx);
        sScore[tid + q * 256] = p;
        sum += p;
    }
    sRed[tid] = sum;
    __syncthreads();
    for (int off = 128; off > 0; off >>= 1) {
        if (tid < off) sRed[tid] += sRed[tid + off];
        __syncthreads();
    }
    const float rden = fast_rcp(sRed[0]);
    __syncthreads();

    const int d = tid & (NS - 1);
    const int half = tid >> 7;
    float acc = 0.0f;
    const float* po = pre_out + ((size_t)b * TXX + half * 256) * NS + d;
    for (int t = 0; t < 256; ++t) acc += sScore[half * 256 + t] * po[(size_t)t * NS];
    sRed[tid] = acc;
    __syncthreads();
    if (tid < NS) ctx[(size_t)b * NS + tid] = (sRed[tid] + sRed[tid + NS]) * rden;
}

// ---------- decoder LSTM cell: z=[ctx|s]@[Wih_p;Whh_p]^T via bf16 WMMA ----------
__global__ __launch_bounds__(256) void k_deccell(
    const float* __restrict__ ctx, float* __restrict__ s, float* __restrict__ c,
    const unsigned short* __restrict__ WihT,   // fragment-major [4 kt][32 nt][32][16]
    const unsigned short* __restrict__ WhhT,
    const float* __restrict__ bp)
{
    __shared__ Frag sA[8 * 32];                     // 8 KB: [ctx | s] 16 x 256 swizzled
    __shared__ __align__(32) float sZ[32 * 256];    // 32 KB fragment-major accum
    const int tid  = threadIdx.x;
    const int lane = tid & 31;
    const int wave = tid >> 5;
    const int b0   = blockIdx.x * 16;

    unsigned short* sAu = (unsigned short*)sA;
    for (int i = tid; i < 16 * 2 * NS; i += 256) {
        int m = i >> 8, k = i & 255;
        float v = (k < NS) ? ctx[(size_t)(b0 + m) * NS + k]
                           : s[(size_t)(b0 + m) * NS + (k - NS)];
        sAu[aidx(m, k)] = f2bf(v);
    }
    __syncthreads();

    const Frag* gWih = (const Frag*)WihT;
    const Frag* gWhh = (const Frag*)WhhT;
    #pragma unroll
    for (int j = 0; j < 4; ++j) {
        const int nt = wave * 4 + j;
        v8f acc;
        const float bv = bp[nt * 16 + (lane & 15)];
        #pragma unroll
        for (int r = 0; r < 8; ++r) acc[r] = bv;
        #pragma unroll
        for (int kt = 0; kt < 8; ++kt) {
            v16bf bf = (kt < 4) ? loadFrag(gWih, kt * 32 + nt, lane)
                                : loadFrag(gWhh, (kt - 4) * 32 + nt, lane);
            acc = wmma_bf16(loadFrag(sA, kt, lane), bf, acc);
        }
        *(v8f*)&sZ[nt * 256 + lane * 8] = acc;
    }
    __syncthreads();

    for (int i = tid; i < 16 * NS; i += 256) {
        int m = i >> 7, jj = i & (NS - 1);
        float zi = sZ[zidx(m, jj)];
        float zf = sZ[zidx(m, NS + jj)];
        float zg = sZ[zidx(m, 2 * NS + jj)];
        float zo = sZ[zidx(m, 3 * NS + jj)];
        size_t gi = (size_t)(b0 + m) * NS + jj;
        float cc = fast_sigmoid(zf) * c[gi] + fast_sigmoid(zi) * fast_tanh(zg);
        float ss = fast_sigmoid(zo) * fast_tanh(cc);
        c[gi] = cc;
        s[gi] = ss;
    }
}

// ---------- output head + softmax over BATCH axis (faithful to reference) ----------
__global__ __launch_bounds__(256) void k_out(
    const float* __restrict__ s, const float* __restrict__ W3,
    const float* __restrict__ b3, float* __restrict__ out, int ty)
{
    __shared__ float sW[2 * NA];
    __shared__ float sRed[256];
    const int tid = threadIdx.x;
    const int v   = blockIdx.x;
    if (tid < 2 * NA) sW[tid] = W3[v * (2 * NA) + tid];
    __syncthreads();
    const float b3v = b3[v];
    float L[4];
    float mx = -1e30f;
    #pragma unroll
    for (int q = 0; q < 4; ++q) {
        int b = tid + q * 256;
        const float* sv = s + (size_t)b * NS;
        float acc = b3v;
        for (int k = 0; k < 2 * NA; ++k) acc += sv[k] * sW[k];
        L[q] = acc;
        mx = fmaxf(mx, acc);
    }
    sRed[tid] = mx; __syncthreads();
    for (int off = 128; off > 0; off >>= 1) {
        if (tid < off) sRed[tid] = fmaxf(sRed[tid], sRed[tid + off]);
        __syncthreads();
    }
    mx = sRed[0]; __syncthreads();
    float sum = 0.0f; float P[4];
    #pragma unroll
    for (int q = 0; q < 4; ++q) { P[q] = __expf(L[q] - mx); sum += P[q]; }
    sRed[tid] = sum; __syncthreads();
    for (int off = 128; off > 0; off >>= 1) {
        if (tid < off) sRed[tid] += sRed[tid + off];
        __syncthreads();
    }
    const float rden = fast_rcp(sRed[0]);
    #pragma unroll
    for (int q = 0; q < 4; ++q) {
        int b = tid + q * 256;
        out[((size_t)b * TYY + ty) * VOUT + v] = P[q] * rden;
    }
}

extern "C" void kernel_launch(void* const* d_in, const int* in_sizes, int n_in,
                              void* d_out, int out_size, void* d_ws, size_t ws_size,
                              hipStream_t stream)
{
    (void)in_sizes; (void)n_in; (void)out_size; (void)ws_size;
    const float* X     = (const float*)d_in[0];
    const float* Wih_f = (const float*)d_in[1];
    const float* Whh_f = (const float*)d_in[2];
    const float* bih_f = (const float*)d_in[3];
    const float* bhh_f = (const float*)d_in[4];
    const float* Wih_b = (const float*)d_in[5];
    const float* Whh_b = (const float*)d_in[6];
    const float* bih_b = (const float*)d_in[7];
    const float* bhh_b = (const float*)d_in[8];
    const float* Wih_p = (const float*)d_in[9];
    const float* Whh_p = (const float*)d_in[10];
    const float* bih_p = (const float*)d_in[11];
    const float* bhh_p = (const float*)d_in[12];
    const float* W1 = (const float*)d_in[13];
    const float* b1 = (const float*)d_in[14];
    const float* W2 = (const float*)d_in[15];
    const float* b2 = (const float*)d_in[16];
    const float* W3 = (const float*)d_in[17];
    const float* b3 = (const float*)d_in[18];
    float* out = (float*)d_out;

    char* ws = (char*)d_ws;
    size_t off = 0;
    auto alloc = [&](size_t bytes) -> void* {
        void* p = ws + off;
        off += (bytes + 255) & ~(size_t)255;
        return p;
    };
    unsigned short* encWihT = (unsigned short*)alloc((size_t)2 * VIN * 4 * NA * 2);
    unsigned short* encWhhT = (unsigned short*)alloc((size_t)2 * NA * 4 * NA * 2);
    float*          encBias = (float*)alloc((size_t)2 * 4 * NA * 4);
    unsigned short* decWihT = (unsigned short*)alloc((size_t)NS * 4 * NS * 2);
    unsigned short* decWhhT = (unsigned short*)alloc((size_t)NS * 4 * NS * 2);
    float*          decBias = (float*)alloc((size_t)4 * NS * 4);
    float*          sState  = (float*)alloc((size_t)B_SZ * NS * 4);
    float*          cState  = (float*)alloc((size_t)B_SZ * NS * 4);
    float*          ctx     = (float*)alloc((size_t)B_SZ * NS * 4);
    float*          preOut  = (float*)alloc((size_t)B_SZ * TXX * NS * 4);
    float*          Ep      = (float*)alloc((size_t)B_SZ * TXX * 10 * 4);

    // weight prep: transpose + bf16 + fragment-major swizzle
    k_convT_sw<<<(4 * NA * VIN + 255) / 256, 256, 0, stream>>>(Wih_f, encWihT, 4 * NA, VIN);
    k_convT_sw<<<(4 * NA * NA + 255) / 256, 256, 0, stream>>>(Whh_f, encWhhT, 4 * NA, NA);
    k_convT_sw<<<(4 * NA * VIN + 255) / 256, 256, 0, stream>>>(Wih_b, encWihT + VIN * 4 * NA, 4 * NA, VIN);
    k_convT_sw<<<(4 * NA * NA + 255) / 256, 256, 0, stream>>>(Whh_b, encWhhT + NA * 4 * NA, 4 * NA, NA);
    k_convT_sw<<<(4 * NS * NS + 255) / 256, 256, 0, stream>>>(Wih_p, decWihT, 4 * NS, NS);
    k_convT_sw<<<(4 * NS * NS + 255) / 256, 256, 0, stream>>>(Whh_p, decWhhT, 4 * NS, NS);
    k_addb<<<1, 256, 0, stream>>>(bih_f, bhh_f, encBias, 4 * NA);
    k_addb<<<1, 256, 0, stream>>>(bih_b, bhh_b, encBias + 4 * NA, 4 * NA);
    k_addb<<<2, 256, 0, stream>>>(bih_p, bhh_p, decBias, 4 * NS);
    k_zero<<<(B_SZ * NS + 255) / 256, 256, 0, stream>>>(sState, B_SZ * NS);
    k_zero<<<(B_SZ * NS + 255) / 256, 256, 0, stream>>>(cState, B_SZ * NS);

    // bidirectional encoder scan (~69 GFLOP on bf16 WMMA, async-double-buffered X)
    dim3 egrid(B_SZ / 16, 2);
    k_encoder<<<egrid, 256, 0, stream>>>(X, encWihT, encWhhT, encBias, preOut);

    // step-invariant attention term, hoisted out of the TY loop
    k_ep<<<(B_SZ * TXX + 255) / 256, 256, 0, stream>>>(preOut, W1, Ep);

    for (int ty = 0; ty < TYY; ++ty) {
        k_attn<<<B_SZ, 256, 0, stream>>>(Ep, preOut, sState, W1, b1, W2, b2, ctx);
        k_deccell<<<B_SZ / 16, 256, 0, stream>>>(ctx, sState, cState, decWihT, decWhhT, decBias);
        k_out<<<VOUT, 256, 0, stream>>>(sState, W3, b3, out, ty);
    }
}
